// VectorQuantizer_8065948582428
// MI455X (gfx1250) — matrix operand order repaired and checked by reading, not compiled, not run on hardware
//
#include <hip/hip_runtime.h>

typedef _Float16 v16h __attribute__((ext_vector_type(16)));
typedef _Float16 v8h  __attribute__((ext_vector_type(8)));
typedef _Float16 v2h  __attribute__((ext_vector_type(2)));
typedef float    v8f  __attribute__((ext_vector_type(8)));

#if defined(__HIP_DEVICE_COMPILE__) && !__has_builtin(__builtin_amdgcn_wmma_f32_16x16x32_f16)
#error "missing __builtin_amdgcn_wmma_f32_16x16x32_f16 on this toolchain"
#endif

#define N_E    1024
#define E_DIM  64
#define ROWS   65536                 // 64*32*32
#define LDSW   36                    // row stride in dwords: 32 data + 4 pad (16B-aligned rows)
#define ZQ_ELEMS (ROWS * E_DIM)      // 4194304
#define LOSS_OFF ZQ_ELEMS
#define IDX_OFF  (ZQ_ELEMS + 1)
// hi codebook + lo codebook + fp32 norms
#define LDS_BYTES (2 * N_E * LDSW * 4 + N_E * 4)   // 299,008 B < 320KB

extern "C" __global__ __launch_bounds__(256, 1)
void vq_wmma_kernel(const float* __restrict__ z,
                    const float* __restrict__ emb,
                    float* __restrict__ out) {
    extern __shared__ char smem[];
    v2h*   ldsHi = (v2h*)smem;                    // [N_E * LDSW] packed f16 hi
    v2h*   ldsLo = ldsHi + N_E * LDSW;            // [N_E * LDSW] packed f16 lo (residual)
    float* snorm = (float*)(ldsLo + N_E * LDSW);  // [N_E] fp32 ||e||^2
    const v8h* ldsHiW = (const v8h*)smem;                       // 16B chunks
    const v8h* ldsLoW = (const v8h*)(smem + N_E * LDSW * 4);

    const int tid = threadIdx.x;

    // ---- stage split-precision codebook into LDS + fp32 norms ----
    for (int n = tid; n < N_E; n += 256) {
        const float* er = emb + n * E_DIM;
        float s = 0.f;
        #pragma unroll
        for (int k2 = 0; k2 < E_DIM / 2; ++k2) {
            float v0 = er[2 * k2], v1 = er[2 * k2 + 1];
            s += v0 * v0 + v1 * v1;
            _Float16 h0 = (_Float16)v0, h1 = (_Float16)v1;
            v2h hi; hi.x = h0; hi.y = h1;
            v2h lo; lo.x = (_Float16)(v0 - (float)h0); lo.y = (_Float16)(v1 - (float)h1);
            ldsHi[n * LDSW + k2] = hi;
            ldsLo[n * LDSW + k2] = lo;
        }
        snorm[n] = s;
    }
    __syncthreads();

    const int wave = tid >> 5;
    const int lane = tid & 31;
    const int l15  = lane & 15;
    const int half = lane >> 4;                    // 0: lanes 0-15, 1: lanes 16-31
    const int r0   = (blockIdx.x * 8 + wave) * 16; // 16 z-rows per wave

    // ---- A panel: 16 rows x 64 K of z, split into f16 hi/lo 16x16x32 fragments.
    // A layout used here (contiguous k, same as B): VGPR j, lane l -> M=l%16; K = 16*half + 2j + {0,1}
    const float* zrow = z + (size_t)(r0 + l15) * E_DIM;
    v16h ahi[2], alo[2];
    #pragma unroll
    for (int f = 0; f < 2; ++f) {
        const int c0 = f * 32 + 16 * half;
        #pragma unroll
        for (int e = 0; e < 16; ++e) {
            const int col = c0 + e;   // 0..63
            float v = zrow[col];
            _Float16 h = (_Float16)v;
            ahi[f][e] = h;
            alo[f][e] = (_Float16)(v - (float)h);
        }
    }

    float best[8];
    int   bidx[8];
    #pragma unroll
    for (int v = 0; v < 8; ++v) { best[v] = __builtin_inff(); bidx[v] = 0; }

    // ---- sweep 64 tiles of 16 codes ----
    for (int nt = 0; nt < N_E / 16; ++nt) {
        const int n = nt * 16 + l15;               // B: lane l -> column N = n
        // B layout used here (32x16, same k order as A): VGPR j, lane l -> N=l%16; K = 16*half + 2j + {0,1}
        // fragment = 8 consecutive dwords at dword (n*LDSW + f*16 + 8*half) -> two b128 loads
        v16h bhi[2], blo[2];
        #pragma unroll
        for (int f = 0; f < 2; ++f) {
            const int w = (n * LDSW + f * 16 + 8 * half) >> 2;   // v8h index (16B units)
            v8h h0 = ldsHiW[w], h1 = ldsHiW[w + 1];
            v8h l0 = ldsLoW[w], l1 = ldsLoW[w + 1];
            bhi[f] = __builtin_shufflevector(h0, h1, 0,1,2,3,4,5,6,7,8,9,10,11,12,13,14,15);
            blo[f] = __builtin_shufflevector(l0, l1, 0,1,2,3,4,5,6,7,8,9,10,11,12,13,14,15);
        }
        // three independent accumulator chains: hi*hi, hi*lo, lo*hi
        v8f c0 = {}, c1 = {}, c2 = {};
        c0 = __builtin_amdgcn_wmma_f32_16x16x32_f16(false, ahi[0], false, bhi[0], (short)0, c0, false, false);
        c1 = __builtin_amdgcn_wmma_f32_16x16x32_f16(false, ahi[0], false, blo[0], (short)0, c1, false, false);
        c2 = __builtin_amdgcn_wmma_f32_16x16x32_f16(false, alo[0], false, bhi[0], (short)0, c2, false, false);
        c0 = __builtin_amdgcn_wmma_f32_16x16x32_f16(false, ahi[1], false, bhi[1], (short)0, c0, false, false);
        c1 = __builtin_amdgcn_wmma_f32_16x16x32_f16(false, ahi[1], false, blo[1], (short)0, c1, false, false);
        c2 = __builtin_amdgcn_wmma_f32_16x16x32_f16(false, alo[1], false, bhi[1], (short)0, c2, false, false);

        const float e2 = snorm[n];
        // C layout: VGPR v, lane l -> M = v + 8*half, N = l%16
        #pragma unroll
        for (int v = 0; v < 8; ++v) {
            float dot = (c0[v] + c1[v]) + c2[v];
            float d = __builtin_fmaf(-2.0f, dot, e2);
            if (d < best[v]) { best[v] = d; bidx[v] = n; }
        }
    }

    // ---- min+argmin across the 16 lanes of each half-wave ----
    #pragma unroll
    for (int v = 0; v < 8; ++v) {
        #pragma unroll
        for (int off = 8; off >= 1; off >>= 1) {
            float ob = __shfl_xor(best[v], off, 32);
            int   oi = __shfl_xor(bidx[v], off, 32);
            if (ob < best[v] || (ob == best[v] && oi < bidx[v])) { best[v] = ob; bidx[v] = oi; }
        }
    }

    // ---- outputs: z_q (straight-through forward == z_q), indices, loss partials ----
    float acc = 0.f;
    #pragma unroll
    for (int v = 0; v < 8; ++v) {
        const int row = r0 + v + 8 * half;
        const int idx = bidx[v];
        if (l15 == 0) out[IDX_OFF + row] = (float)idx;
        const float* erow = emb + (size_t)idx * E_DIM;   // exact fp32 codebook (L2-resident)
        #pragma unroll
        for (int j = 0; j < 4; ++j) {
            const int col = l15 + 16 * j;
            const float e  = erow[col];
            const float zv = z[(size_t)row * E_DIM + col];
            out[(size_t)row * E_DIM + col] = e;
            const float dd = e - zv;
            acc += dd * dd;
        }
    }
    #pragma unroll
    for (int off = 16; off >= 1; off >>= 1) acc += __shfl_xor(acc, off, 32);
    // loss = mean((zq-z)^2) + mean((zq-z)^2) = 2*sum/N
    if (lane == 0) atomicAdd(out + LOSS_OFF, acc * (2.0f / (float)ZQ_ELEMS));
}

extern "C" void kernel_launch(void* const* d_in, const int* in_sizes, int n_in,
                              void* d_out, int out_size, void* d_ws, size_t ws_size,
                              hipStream_t stream) {
    const float* z   = (const float*)d_in[0];   // [65536, 64]
    const float* emb = (const float*)d_in[1];   // [1024, 64]
    float* out = (float*)d_out;                 // [zq 4194304 | loss 1 | idx 65536]

    // zero the loss accumulator (graph-capturable memset node)
    (void)hipMemsetAsync(out + LOSS_OFF, 0, sizeof(float), stream);

    // 512 blocks x 256 threads: each of the 4096 waves owns 16 rows
    dim3 grid(ROWS / (8 * 16));
    dim3 block(256);
    vq_wmma_kernel<<<grid, block, LDS_BYTES, stream>>>(z, emb, out);
}
